// MultiHeadAttention_65936337928878
// MI455X (gfx1250) — matrix operand order repaired and checked by reading, not compiled, not run on hardware
//
#include <hip/hip_runtime.h>

typedef __attribute__((ext_vector_type(16))) _Float16 v16h;
typedef __attribute__((ext_vector_type(8)))  _Float16 v8h;
typedef __attribute__((ext_vector_type(8)))  float    v8f;

#define B_   32
#define S_   600
#define SP_  608      // S padded to multiple of 32
#define D_   768
#define DH_  48
#define NH_  16
#define BH_  (B_ * NH_)        // 512
#define QT_  (SP_ / 16)        // 38 query tiles
#define SCALE_ 0.14433756729740643f   // 48^-0.5

// ---------------------------------------------------------------------------
// Kernel 1: factored TLE for Q,K,V. One block (128 thr) per token.
// Writes f16 Qh/Kh as [bh][SP_][48] (rows >=600 left untouched; masked later)
// and V transposed: Vt [bh][48][SP_].
// ---------------------------------------------------------------------------
__global__ __launch_bounds__(128) void tle_qkv_kernel(
    const float* __restrict__ x,
    const float* __restrict__ wq1, const float* __restrict__ wq2,
    const float* __restrict__ wq3, const float* __restrict__ bq,
    const float* __restrict__ wk1, const float* __restrict__ wk2,
    const float* __restrict__ wk3, const float* __restrict__ bk,
    const float* __restrict__ wv1, const float* __restrict__ wv2,
    const float* __restrict__ wv3, const float* __restrict__ bv,
    _Float16* __restrict__ Qh, _Float16* __restrict__ Kh,
    _Float16* __restrict__ Vt)
{
    __shared__ float xs[D_], y1[D_], y2[D_];
    const int t = blockIdx.x;
    const int b = t / S_;
    const int s = t % S_;

    const float* xp = x + (size_t)t * D_;
    for (int e = threadIdx.x; e < D_; e += 128) xs[e] = xp[e];
    __syncthreads();

    const float* W1[3] = {wq1, wk1, wv1};
    const float* W2[3] = {wq2, wk2, wv2};
    const float* W3[3] = {wq3, wk3, wv3};
    const float* BI[3] = {bq,  bk,  bv };

#pragma unroll
    for (int proj = 0; proj < 3; ++proj) {
        const float* w1 = W1[proj];
        const float* w2 = W2[proj];
        const float* w3 = W3[proj];
        const float* bi = BI[proj];

        // stage A: contract mode-1 (i, 8)
        for (int o = threadIdx.x; o < D_; o += 128) {
            int a = o / 96, rem = o % 96;
            float acc = 0.0f;
#pragma unroll
            for (int i = 0; i < 8; ++i) acc += xs[i * 96 + rem] * w1[a * 8 + i];
            y1[o] = acc;
        }
        __syncthreads();
        // stage B: contract mode-2 (j, 8)
        for (int o = threadIdx.x; o < D_; o += 128) {
            int a = o / 96, c = (o % 96) / 12, k = o % 12;
            float acc = 0.0f;
#pragma unroll
            for (int j = 0; j < 8; ++j) acc += y1[a * 96 + j * 12 + k] * w2[c * 8 + j];
            y2[o] = acc;
        }
        __syncthreads();
        // stage C: contract mode-3 (k, 12), add bias, scatter to head layout
        for (int o = threadIdx.x; o < D_; o += 128) {
            int a = o / 96, c = (o % 96) / 12, d = o % 12;
            float acc = bi[o];
#pragma unroll
            for (int k = 0; k < 12; ++k) acc += y2[a * 96 + c * 12 + k] * w3[d * 12 + k];
            // a = x*2+h1, c = y*2+h2, d = z*4+h3
            int h1 = a & 1, xx = a >> 1;
            int h2 = c & 1, yy = c >> 1;
            int h3 = d & 3, zz = d >> 2;
            int bh = b * 16 + h1 * 8 + h2 * 4 + h3;
            int dh = xx * 12 + yy * 3 + zz;
            _Float16 hv = (_Float16)acc;
            if (proj == 0)      Qh[((size_t)bh * SP_ + s) * DH_ + dh] = hv;
            else if (proj == 1) Kh[((size_t)bh * SP_ + s) * DH_ + dh] = hv;
            else                Vt[((size_t)bh * DH_ + dh) * SP_ + s] = hv;
        }
        __syncthreads();
    }
}

// ---------------------------------------------------------------------------
// Kernel 2: flash attention, one wave32 per (bh, 16-query tile).
// Computes S^T = K * Q^T (keys on rows -> softmax reduction = xor-16 shuffle)
// and O^T = V^T * P^T, all via v_wmma_f32_16x16x32_f16.
// ---------------------------------------------------------------------------
__global__ __launch_bounds__(32) void attn_kernel(
    const _Float16* __restrict__ Qh, const _Float16* __restrict__ Kh,
    const _Float16* __restrict__ Vt, float* __restrict__ Oh)
{
    const int bh    = blockIdx.x / QT_;
    const int qt    = blockIdx.x % QT_;
    const int qbase = qt * 16;
    const int lane  = threadIdx.x;
    const int n     = lane & 15;   // query column / matrix row-within-tile
    const int hi    = lane >> 4;   // lane half

    const _Float16* Qp = Qh + (size_t)bh * SP_ * DH_;
    const _Float16* Kp = Kh + (size_t)bh * SP_ * DH_;
    const _Float16* Vp = Vt + (size_t)bh * DH_ * SP_;

    union V16 { v16h v; v8h h[2]; };
    const v8h  zero8  = {};
    const v16h zero16 = {};
    const v8f  zc     = {};

    // Q as B-fragments of S^T (col n = query qbase+n, K index = head dim)
    const int qrow = qbase + n;                       // may be >=600 (padded mem)
    V16 uq;
    uq.h[0] = *(const v8h*)(Qp + (size_t)qrow * DH_ + hi * 8);
    uq.h[1] = *(const v8h*)(Qp + (size_t)qrow * DH_ + 16 + hi * 8);
    v16h bq1 = uq.v;                                  // dims 0..31
    uq.h[0] = *(const v8h*)(Qp + (size_t)qrow * DH_ + 32 + hi * 8);
    uq.h[1] = zero8;                                  // dims 48..63 are zero
    v16h bq2 = uq.v;                                  // dims 32..47

    v8f o0 = {}, o1 = {}, o2 = {};
    float m_i = -1e30f, l_i = 0.0f;

    for (int kb = 0; kb < SP_; kb += 32) {
        // --- K tile A-fragments (rows = keys, K = head dims, padded to 64) ---
        const _Float16* kp0 = Kp + (size_t)(kb + n) * DH_;
        const _Float16* kp1 = Kp + (size_t)(kb + 16 + n) * DH_;
        V16 ua;
        ua.h[0] = *(const v8h*)(kp0 + hi * 8);
        ua.h[1] = *(const v8h*)(kp0 + 16 + hi * 8);
        v16h a1_0 = ua.v;                       // dims 0..31
        ua.h[0] = *(const v8h*)(kp0 + 32 + hi * 8);
        ua.h[1] = zero8;
        v16h a2_0 = ua.v;                       // dims 32..47 (+zeros)
        ua.h[0] = *(const v8h*)(kp1 + hi * 8);
        ua.h[1] = *(const v8h*)(kp1 + 16 + hi * 8);
        v16h a1_1 = ua.v;
        ua.h[0] = *(const v8h*)(kp1 + 32 + hi * 8);
        ua.h[1] = zero8;
        v16h a2_1 = ua.v;

        // --- S^T tiles: T0 = keys kb..kb+15, T1 = keys kb+16..kb+31 ---
        v8f s0 = __builtin_amdgcn_wmma_f32_16x16x32_f16(false, a1_0, false, bq1,
                                                        (short)0, zc, false, false);
        s0     = __builtin_amdgcn_wmma_f32_16x16x32_f16(false, a2_0, false, bq2,
                                                        (short)0, s0, false, false);
        v8f s1 = __builtin_amdgcn_wmma_f32_16x16x32_f16(false, a1_1, false, bq1,
                                                        (short)0, zc, false, false);
        s1     = __builtin_amdgcn_wmma_f32_16x16x32_f16(false, a2_1, false, bq2,
                                                        (short)0, s1, false, false);

        // --- masked, scaled scores; online softmax over this 32-key block ---
        float t0[8], t1[8];
        float tmax = -1e30f;
#pragma unroll
        for (int r = 0; r < 8; ++r) {
            int k0 = kb + r + 8 * hi;
            int k1 = kb + 16 + r + 8 * hi;
            t0[r] = (k0 < S_) ? s0[r] * SCALE_ : -1e30f;
            t1[r] = (k1 < S_) ? s1[r] * SCALE_ : -1e30f;
            tmax = fmaxf(tmax, fmaxf(t0[r], t1[r]));
        }
        tmax = fmaxf(tmax, __shfl_xor(tmax, 16, 32));   // column (=query) max
        float m_new = fmaxf(m_i, tmax);
        float alpha = __expf(m_i - m_new);

        float p0[8], p1[8];
        float lsum = 0.0f;
#pragma unroll
        for (int r = 0; r < 8; ++r) {
            p0[r] = __expf(t0[r] - m_new);    // masked entries underflow to 0
            p1[r] = __expf(t1[r] - m_new);
            lsum += p0[r] + p1[r];
        }
        lsum += __shfl_xor(lsum, 16, 32);
        l_i = alpha * l_i + lsum;
        m_i = m_new;

#pragma unroll
        for (int r = 0; r < 8; ++r) { o0[r] *= alpha; o1[r] *= alpha; o2[r] *= alpha; }

        // --- build B-fragment of P^T (K = 32 keys, col n = query) ---
        v16h pb = {};
#pragma unroll
        for (int r = 0; r < 8; ++r) {
            float x0 = __shfl_xor(p0[r], 16, 32);
            float x1 = __shfl_xor(p1[r], 16, 32);
            pb[r]     = (_Float16)p0[r];
            pb[8 + r] = (_Float16)p1[r];
        }

        // --- O^T += V^T * P^T, three 16-dim tiles ---
#pragma unroll
        for (int dt = 0; dt < 3; ++dt) {
            const _Float16* vp = Vp + (size_t)(dt * 16 + n) * SP_ + kb;
            V16 uv;
            uv.h[0] = *(const v8h*)(vp + hi * 8);
            uv.h[1] = *(const v8h*)(vp + 16 + hi * 8);
            if (dt == 0)
                o0 = __builtin_amdgcn_wmma_f32_16x16x32_f16(false, uv.v, false, pb,
                                                            (short)0, o0, false, false);
            else if (dt == 1)
                o1 = __builtin_amdgcn_wmma_f32_16x16x32_f16(false, uv.v, false, pb,
                                                            (short)0, o1, false, false);
            else
                o2 = __builtin_amdgcn_wmma_f32_16x16x32_f16(false, uv.v, false, pb,
                                                            (short)0, o2, false, false);
        }
    }

    // --- epilogue: divide by softmax denominator, store O[bh][query][48] ---
    const float inv = 1.0f / l_i;
    if (qrow < S_) {
        float* op = Oh + ((size_t)bh * S_ + qrow) * DH_ + 8 * hi;
#pragma unroll
        for (int r = 0; r < 8; ++r) {
            op[r]      = o0[r] * inv;
            op[16 + r] = o1[r] * inv;
            op[32 + r] = o2[r] * inv;
        }
    }
}

// ---------------------------------------------------------------------------
// Kernel 3: merge heads + output TLE. One block (128 thr) per token.
// ---------------------------------------------------------------------------
__global__ __launch_bounds__(128) void tle_out_kernel(
    const float* __restrict__ Oh,
    const float* __restrict__ wo1, const float* __restrict__ wo2,
    const float* __restrict__ wo3, const float* __restrict__ bo,
    float* __restrict__ out)
{
    __shared__ float xs[D_], y1[D_], y2[D_];
    const int t = blockIdx.x;
    const int b = t / S_;
    const int s = t % S_;

    // gather merged-head vector for this token
    for (int o = threadIdx.x; o < D_; o += 128) {
        int a = o / 96, c = (o % 96) / 12, d = o % 12;
        int bh = b * 16 + (a & 1) * 8 + (c & 1) * 4 + (d & 3);
        int dh = (a >> 1) * 12 + (c >> 1) * 3 + (d >> 2);
        xs[o] = Oh[((size_t)bh * S_ + s) * DH_ + dh];
    }
    __syncthreads();

    for (int o = threadIdx.x; o < D_; o += 128) {
        int a = o / 96, rem = o % 96;
        float acc = 0.0f;
#pragma unroll
        for (int i = 0; i < 8; ++i) acc += xs[i * 96 + rem] * wo1[a * 8 + i];
        y1[o] = acc;
    }
    __syncthreads();
    for (int o = threadIdx.x; o < D_; o += 128) {
        int a = o / 96, c = (o % 96) / 12, k = o % 12;
        float acc = 0.0f;
#pragma unroll
        for (int j = 0; j < 8; ++j) acc += y1[a * 96 + j * 12 + k] * wo2[c * 8 + j];
        y2[o] = acc;
    }
    __syncthreads();
    float* op = out + (size_t)t * D_;
    for (int o = threadIdx.x; o < D_; o += 128) {
        int a = o / 96, c = (o % 96) / 12, d = o % 12;
        float acc = bo[o];
#pragma unroll
        for (int k = 0; k < 12; ++k) acc += y2[a * 96 + c * 12 + k] * wo3[d * 12 + k];
        op[o] = acc;
    }
}

// ---------------------------------------------------------------------------
extern "C" void kernel_launch(void* const* d_in, const int* in_sizes, int n_in,
                              void* d_out, int out_size, void* d_ws, size_t ws_size,
                              hipStream_t stream)
{
    (void)in_sizes; (void)n_in; (void)out_size; (void)ws_size;
    const float* x   = (const float*)d_in[0];
    const float* wq1 = (const float*)d_in[1];
    const float* wq2 = (const float*)d_in[2];
    const float* wq3 = (const float*)d_in[3];
    const float* bq  = (const float*)d_in[4];
    const float* wk1 = (const float*)d_in[5];
    const float* wk2 = (const float*)d_in[6];
    const float* wk3 = (const float*)d_in[7];
    const float* bk  = (const float*)d_in[8];
    const float* wv1 = (const float*)d_in[9];
    const float* wv2 = (const float*)d_in[10];
    const float* wv3 = (const float*)d_in[11];
    const float* bv  = (const float*)d_in[12];
    const float* wo1 = (const float*)d_in[13];
    const float* wo2 = (const float*)d_in[14];
    const float* wo3 = (const float*)d_in[15];
    const float* bo  = (const float*)d_in[16];
    float* out = (float*)d_out;

    // workspace layout: Qh | Kh | Vt (f16) | Oh (f32)  ~= 148.6 MB
    const size_t qk = (size_t)BH_ * SP_ * DH_;   // elements per f16 buffer
    _Float16* Qh = (_Float16*)d_ws;
    _Float16* Kh = Qh + qk;
    _Float16* Vt = Kh + qk;
    float*    Oh = (float*)(Vt + qk);

    tle_qkv_kernel<<<B_ * S_, 128, 0, stream>>>(
        x, wq1, wq2, wq3, bq, wk1, wk2, wk3, bk, wv1, wv2, wv3, bv, Qh, Kh, Vt);
    attn_kernel<<<BH_ * QT_, 32, 0, stream>>>(Qh, Kh, Vt, Oh);
    tle_out_kernel<<<B_ * S_, 128, 0, stream>>>(Oh, wo1, wo2, wo3, bo, out);
}
